// MoEBlock_30502857736769
// MI455X (gfx1250) — compile-verified, matrix operand order checked by reading, not run on hardware
//
#include <hip/hip_runtime.h>
#include <stdint.h>

typedef __attribute__((ext_vector_type(16))) __bf16 v16bf;
typedef __attribute__((ext_vector_type(8)))  __bf16 v8bf;
typedef __attribute__((ext_vector_type(8)))  float  v8f;
typedef __attribute__((ext_vector_type(4)))  int    v4i;

// address-space qualified element types for the async-copy builtin
typedef __attribute__((address_space(1))) v4i v4i_g;   // global
typedef __attribute__((address_space(3))) v4i v4i_l;   // LDS

#define D_MODEL 1024
#define D_FF    4096
#define RANK    16
#define M_TOK   16384        // B * S = 8 * 2048
#define KP      32           // lora K padded 16 -> 32 (zero fill)
#define LDS_STRIDE 40        // 32 elems + 8 pad (80B rows, 16B aligned)

#if defined(__HIP_DEVICE_COMPILE__) && \
    __has_builtin(__builtin_amdgcn_global_load_async_to_lds_b128)
#define HAVE_ASYNC 1
#else
#define HAVE_ASYNC 0
#endif

__device__ __forceinline__ v16bf cat8(v8bf lo, v8bf hi) {
  return __builtin_shufflevector(lo, hi, 0,1,2,3,4,5,6,7,8,9,10,11,12,13,14,15);
}

__device__ __forceinline__ v8f wmma_bf16(v16bf a, v16bf b, v8f c) {
  return __builtin_amdgcn_wmma_f32_16x16x32_bf16(false, a, false, b, (short)0, c,
                                                 false, false);
}

// Async 16B copy global -> LDS (ASYNCcnt); fallback: direct store.
__device__ __forceinline__ void async_cp16(const __bf16* gsrc, __bf16* ldst) {
#if HAVE_ASYNC
  __builtin_amdgcn_global_load_async_to_lds_b128(
      (v4i_g*)(uintptr_t)gsrc, (v4i_l*)ldst, 0, 0);
#else
  *(v8bf*)ldst = *(const v8bf*)gsrc;
#endif
}

__device__ __forceinline__ void async_wait0() {
#if HAVE_ASYNC
#if __has_builtin(__builtin_amdgcn_s_wait_asynccnt)
  __builtin_amdgcn_s_wait_asynccnt(0);
#else
  asm volatile("s_wait_asynccnt 0x0" ::: "memory");
#endif
#endif
}

// Load an A-layout fragment (16x32 bf16) from a row-major [rows x ldK] matrix:
// lane l -> row (l&15), K chunks at ((l>>4)*8) and ((l>>4)*8 + 16).
__device__ __forceinline__ v16bf frag_global(const __bf16* __restrict__ base,
                                             int ldK, int lane) {
  const __bf16* p = base + (size_t)(lane & 15) * ldK + ((lane >> 4) << 3);
  return cat8(*(const v8bf*)p, *(const v8bf*)(p + 16));
}

// ---------------------------------------------------------------------------
// Kernel 1: convert wi/wo to bf16; build padded per-expert LoRA-B [2][F][32]
// ---------------------------------------------------------------------------
__global__ __launch_bounds__(256) void prep_weights(
    const float* __restrict__ wi, const float* __restrict__ wo,
    const float* __restrict__ loraB, const int* __restrict__ idx,
    __bf16* __restrict__ wi_b, __bf16* __restrict__ wo_b,
    __bf16* __restrict__ Bl) {
  const size_t NW = (size_t)D_FF * D_MODEL;            // 4M elems each
  const size_t NBL = 2ull * D_FF * KP;                 // 256K elems
  size_t i = (size_t)blockIdx.x * blockDim.x + threadIdx.x;
  if (i < NW) {
    wi_b[i] = (__bf16)wi[i];
  } else if (i < 2 * NW) {
    size_t j = i - NW;
    wo_b[j] = (__bf16)wo[j];
  } else if (i < 2 * NW + NBL) {
    size_t j = i - 2 * NW;
    int k = (int)(j / ((size_t)D_FF * KP));
    size_t rem = j - (size_t)k * D_FF * KP;
    int f = (int)(rem / KP);
    int r = (int)(rem % KP);
    float v = 0.0f;
    if (r < RANK) v = loraB[(size_t)idx[k] * D_FF * RANK + (size_t)f * RANK + r];
    Bl[j] = (__bf16)v;
  }
}

// ---------------------------------------------------------------------------
// Kernel 2: per token row: h fp32 -> bf16, and t_k[m,:] = h[m,:] @ A_k^T
// t stored padded [2][M_TOK][KP] bf16 (r>=16 zero).
// ---------------------------------------------------------------------------
__global__ __launch_bounds__(256) void prep_tokens(
    const float* __restrict__ h, const float* __restrict__ loraA,
    const int* __restrict__ idx, __bf16* __restrict__ h_b,
    __bf16* __restrict__ t01) {
  __shared__ float hrow[D_MODEL];
  __shared__ float partial[32][8];
  const int m = blockIdx.x;
  const float* hp = h + (size_t)m * D_MODEL;
  for (int d = threadIdx.x; d < D_MODEL; d += 256) {
    float v = hp[d];
    hrow[d] = v;
    h_b[(size_t)m * D_MODEL + d] = (__bf16)v;
  }
  __syncthreads();
  // 32 dot-products (2 experts x 16 ranks), 8 segments of 128 each
  const int rid = threadIdx.x & 31;     // expert*16 + r
  const int seg = threadIdx.x >> 5;     // 0..7
  const int k = rid >> 4, r = rid & 15;
  const float* Arow =
      loraA + (size_t)idx[k] * RANK * D_MODEL + (size_t)r * D_MODEL;
  float s = 0.0f;
  const int d0 = seg * 128;
  #pragma unroll 4
  for (int d = d0; d < d0 + 128; ++d) s += hrow[d] * Arow[d];
  partial[rid][seg] = s;
  __syncthreads();
  if (threadIdx.x < 32) {
    float tot = 0.0f;
    #pragma unroll
    for (int j = 0; j < 8; ++j) tot += partial[threadIdx.x][j];
    const int kk = threadIdx.x >> 4, rr = threadIdx.x & 15;
    __bf16* trow = t01 + ((size_t)kk * M_TOK + m) * KP;
    trow[rr] = (__bf16)tot;
    trow[rr + 16] = (__bf16)0.0f;     // zero K padding
  }
}

// ---------------------------------------------------------------------------
// Tiled bf16 WMMA GEMM: C[M,N] = A[M,K] * Bt[N,K]^T
// Workgroup: 256 threads (8 waves), 128x128 tile; wave = 32x64 subtile.
// Mainloop: double-buffered LDS, tiles staged with GLOBAL_LOAD_ASYNC_TO_LDS
// (ASYNCcnt) overlapped with WMMA; one barrier per K-step.
// FUSE=1: epilogue adds LoRA (2 extra WMMAs/frag), relu, routing mix -> bf16
// FUSE=0: epilogue stores f32 accumulators
// ---------------------------------------------------------------------------
template <int FUSE>
__global__ __launch_bounds__(256) void moe_gemm(
    const __bf16* __restrict__ A, const __bf16* __restrict__ Bt,
    int M, int N, int K,
    const __bf16* __restrict__ t01,  // [2][M][KP]     (FUSE)
    const __bf16* __restrict__ Bl,   // [2][N][KP]     (FUSE)
    const float* __restrict__ vals,  // [2]            (FUSE)
    __bf16* __restrict__ outb,       // [M][N]         (FUSE)
    float* __restrict__ outf) {      // [M][N]         (!FUSE)
  __shared__ __bf16 Als[2][128 * LDS_STRIDE];
  __shared__ __bf16 Bls[2][128 * LDS_STRIDE];

  const int tid = threadIdx.x;
  const int lane = tid & 31;
  const int wave = tid >> 5;
  const int wm = wave >> 1;          // 0..3: 32-row band
  const int wn = wave & 1;           // 0..1: 64-col band
  const int m0 = blockIdx.y << 7;
  const int n0 = blockIdx.x << 7;
  const int lr = lane & 15;
  const int lc = (lane >> 4) << 3;   // 0 or 8

  v8f acc[2][4] = {};

  // tile staging slots: each thread copies 2x16B for A and 2x16B for B
  const int ldr = tid >> 2;          // 0..63
  const int ldc = (tid & 3) << 3;    // 0,8,16,24
  const __bf16* Ag = A + (size_t)(m0 + ldr) * K + ldc;
  const __bf16* Bg = Bt + (size_t)(n0 + ldr) * K + ldc;

  auto issue_tile = [&](int k0, int buf) {
    #pragma unroll
    for (int i = 0; i < 2; ++i) {
      const int row = ldr + (i << 6);
      async_cp16(Ag + (size_t)(i << 6) * K + k0,
                 &Als[buf][row * LDS_STRIDE + ldc]);
      async_cp16(Bg + (size_t)(i << 6) * K + k0,
                 &Bls[buf][row * LDS_STRIDE + ldc]);
    }
  };

  const int nsteps = K >> 5;
  issue_tile(0, 0);

  for (int s = 0; s < nsteps; ++s) {
    const int buf = s & 1;
    async_wait0();                 // this wave's writes to `buf` have landed
    __syncthreads();               // all waves: `buf` ready, buf^1 free
    if (s + 1 < nsteps) issue_tile((s + 1) << 5, buf ^ 1);  // overlaps WMMA

    v16bf af[2], bfm[4];
    #pragma unroll
    for (int mi = 0; mi < 2; ++mi) {
      const __bf16* p = &Als[buf][(wm * 32 + mi * 16 + lr) * LDS_STRIDE + lc];
      af[mi] = cat8(*(const v8bf*)p, *(const v8bf*)(p + 16));
    }
    #pragma unroll
    for (int ni = 0; ni < 4; ++ni) {
      const __bf16* p = &Bls[buf][(wn * 64 + ni * 16 + lr) * LDS_STRIDE + lc];
      bfm[ni] = cat8(*(const v8bf*)p, *(const v8bf*)(p + 16));
    }
    #pragma unroll
    for (int mi = 0; mi < 2; ++mi)
      #pragma unroll
      for (int ni = 0; ni < 4; ++ni)
        acc[mi][ni] = wmma_bf16(af[mi], bfm[ni], acc[mi][ni]);
  }

  if (FUSE) {
    const float v0 = vals[0], v1 = vals[1];
    v16bf tf[2][2], blf[2][4];
    #pragma unroll
    for (int mi = 0; mi < 2; ++mi) {
      const int m = m0 + wm * 32 + mi * 16;
      tf[0][mi] = frag_global(t01 + (size_t)m * KP, KP, lane);
      tf[1][mi] = frag_global(t01 + ((size_t)M_TOK + m) * KP, KP, lane);
    }
    #pragma unroll
    for (int ni = 0; ni < 4; ++ni) {
      const int n = n0 + wn * 64 + ni * 16;
      blf[0][ni] = frag_global(Bl + (size_t)n * KP, KP, lane);
      blf[1][ni] = frag_global(Bl + ((size_t)N + n) * KP, KP, lane);
    }
    #pragma unroll
    for (int mi = 0; mi < 2; ++mi) {
      #pragma unroll
      for (int ni = 0; ni < 4; ++ni) {
        v8f a0 = wmma_bf16(tf[0][mi], blf[0][ni], acc[mi][ni]);
        v8f a1 = wmma_bf16(tf[1][mi], blf[1][ni], acc[mi][ni]);
        const int mbase = m0 + wm * 32 + mi * 16 + ((lane >> 4) << 3);
        const int nn = n0 + wn * 64 + ni * 16 + (lane & 15);
        #pragma unroll
        for (int j = 0; j < 8; ++j) {
          float r = v0 * fmaxf((float)a0[j], 0.0f) +
                    v1 * fmaxf((float)a1[j], 0.0f);
          outb[(size_t)(mbase + j) * N + nn] = (__bf16)r;
        }
      }
    }
  } else {
    #pragma unroll
    for (int mi = 0; mi < 2; ++mi) {
      #pragma unroll
      for (int ni = 0; ni < 4; ++ni) {
        const int mbase = m0 + wm * 32 + mi * 16 + ((lane >> 4) << 3);
        const int nn = n0 + wn * 64 + ni * 16 + (lane & 15);
        #pragma unroll
        for (int j = 0; j < 8; ++j)
          outf[(size_t)(mbase + j) * N + nn] = acc[mi][ni][j];
      }
    }
  }
}

// ---------------------------------------------------------------------------
extern "C" void kernel_launch(void* const* d_in, const int* in_sizes, int n_in,
                              void* d_out, int out_size, void* d_ws,
                              size_t ws_size, hipStream_t stream) {
  (void)in_sizes; (void)n_in; (void)out_size; (void)ws_size;
  const float* h   = (const float*)d_in[0];
  const float* wi  = (const float*)d_in[1];
  const float* wo  = (const float*)d_in[2];
  const float* lA  = (const float*)d_in[3];
  const float* lB  = (const float*)d_in[4];
  const int*   idx = (const int*)d_in[5];
  const float* val = (const float*)d_in[6];
  float* out = (float*)d_out;

  char* ws = (char*)d_ws;
  size_t off = 0;
  auto alloc = [&](size_t bytes) -> void* {
    void* p = ws + off;
    off = (off + bytes + 255) & ~(size_t)255;
    return p;
  };
  __bf16* h_b  = (__bf16*)alloc((size_t)M_TOK * D_MODEL * 2);   // 32 MB
  __bf16* wi_b = (__bf16*)alloc((size_t)D_FF * D_MODEL * 2);    //  8 MB
  __bf16* wo_b = (__bf16*)alloc((size_t)D_MODEL * D_FF * 2);    //  8 MB
  __bf16* Bl   = (__bf16*)alloc(2ull * D_FF * KP * 2);          // .5 MB
  __bf16* t01  = (__bf16*)alloc(2ull * M_TOK * KP * 2);         //  2 MB
  __bf16* comb = (__bf16*)alloc((size_t)M_TOK * D_FF * 2);      // 128 MB

  const size_t NW = (size_t)D_FF * D_MODEL;
  const size_t prep_total = 2 * NW + 2ull * D_FF * KP;
  prep_weights<<<(unsigned)((prep_total + 255) / 256), 256, 0, stream>>>(
      wi, wo, lB, idx, wi_b, wo_b, Bl);

  prep_tokens<<<M_TOK, 256, 0, stream>>>(h, lA, idx, h_b, t01);

  // Pass 2: combined = v0*relu(h@wi^T + lora0) + v1*relu(h@wi^T + lora1)
  moe_gemm<1><<<dim3(D_FF / 128, M_TOK / 128), 256, 0, stream>>>(
      h_b, wi_b, M_TOK, D_FF, D_MODEL, t01, Bl, val, comb, nullptr);

  // Pass 3: out = combined @ wo^T  (wo [D,F] row-major is Bt directly)
  moe_gemm<0><<<dim3(D_MODEL / 128, M_TOK / 128), 256, 0, stream>>>(
      comb, wo_b, M_TOK, D_MODEL, D_FF, nullptr, nullptr, nullptr, nullptr,
      out);
}
